// DownBlock_17222818857395
// MI455X (gfx1250) — 8-bit matrix operand order repaired and checked by reading, compiled, not run on hardware
//
#include <hip/hip_runtime.h>

typedef int   v4i  __attribute__((ext_vector_type(4)));
typedef int   v16i __attribute__((ext_vector_type(16)));
typedef float v8f  __attribute__((ext_vector_type(8)));

typedef __attribute__((address_space(1))) v4i gbl_v4i;
typedef __attribute__((address_space(3))) v4i lds_v4i;

#if defined(__has_builtin)
#if __has_builtin(__builtin_amdgcn_global_load_async_to_lds_b128) && \
    __has_builtin(__builtin_amdgcn_s_wait_asynccnt)
#define USE_ASYNC_LDS 1
#endif
#endif
#ifndef USE_ASYNC_LDS
#define USE_ASYNC_LDS 0
#endif

namespace {
constexpr int kB = 64, kCin = 512, kCout = 256, kH = 32, kW = 32;
constexpr int kHWc = kH * kW;
constexpr int kNPix = kB * kHWc;
constexpr int kK = kCin * 9;
constexpr size_t kXbBytes = (size_t)kNPix * kCin;
constexpr size_t kWbBytes = (size_t)kCout * kK;
constexpr size_t kConvBytes = (size_t)kNPix * kCout * 4;

constexpr size_t OFF_XB    = 0;
constexpr size_t OFF_WB    = OFF_XB + kXbBytes;
constexpr size_t OFF_CONV  = OFF_WB + kWbBytes;
constexpr size_t OFF_SUM   = OFF_CONV + kConvBytes;
constexpr size_t OFF_SQ    = OFF_SUM + 256 * sizeof(float);
constexpr size_t OFF_SCALE = OFF_SQ  + 256 * sizeof(float);
constexpr size_t OFF_BIAS  = OFF_SCALE + 256 * sizeof(float);

constexpr unsigned char FP8_P1 = 0x38;
constexpr unsigned char FP8_M1 = 0xB8;

constexpr int kActCols = 34;
constexpr int kActBytes = 3 * kActCols * 512;
constexpr int kActChunks = kActBytes / 16;
}

__global__ __launch_bounds__(256) void bin_x_kernel(const float* __restrict__ x,
                                                    unsigned char* __restrict__ xb) {
  int o = blockIdx.x * 256 + threadIdx.x;
  int c = o & (kCin - 1);
  int pix = o >> 9;
  int b = pix >> 10;
  int hw = pix & (kHWc - 1);
  float v = x[(((size_t)b * kCin + c) << 10) + hw];
  xb[o] = (v >= 0.f) ? FP8_P1 : FP8_M1;
}

__global__ __launch_bounds__(256) void bin_w_kernel(const float* __restrict__ Wt,
                                                    unsigned char* __restrict__ wb) {
  int o = blockIdx.x * 256 + threadIdx.x;
  int co = o / kK;
  int k = o - co * kK;
  int pos = k >> 9;
  int ci = k & (kCin - 1);
  float v = Wt[((size_t)(co * kCin + ci)) * 9 + pos];
  wb[o] = (v >= 0.f) ? FP8_P1 : FP8_M1;
}

__global__ void zero_stats_kernel(float* __restrict__ stats) {
  stats[threadIdx.x] = 0.f;
}

__global__ __launch_bounds__(256) void conv_wmma_kernel(
    const unsigned char* __restrict__ xb, const unsigned char* __restrict__ wb,
    float* __restrict__ convout, float* __restrict__ chansum, float* __restrict__ chansq) {
  __shared__ __align__(16) unsigned char act[kActBytes];
  __shared__ float lsum[128], lsq[128];

  const int tid  = threadIdx.x;
  const int wave = tid >> 5, lane = tid & 31;
  const int col = lane & 15, khalf = lane >> 4, M = lane & 15;

  const int p0 = blockIdx.x * 32;
  const int b = p0 >> 10, h = (p0 >> 5) & 31;
  const int cobase = blockIdx.y * 128 + wave * 16;

  if (tid < 128) { lsum[tid] = 0.f; lsq[tid] = 0.f; }

  for (int q = tid; q < kActChunks; q += 256) {
    const int cell = q >> 5;
    const int ci16 = (q & 31) << 4;
    const int r = cell / kActCols, c = cell - r * kActCols;
    const int hp = h - 1 + r, wp = c - 1;
    unsigned char* dst = act + cell * 512 + ci16;
    if ((hp >= 0) & (hp < kH) & (wp >= 0) & (wp < kW)) {
      const unsigned char* src = xb + (((size_t)((b * kH + hp) * kW + wp)) << 9) + ci16;
#if USE_ASYNC_LDS
      __builtin_amdgcn_global_load_async_to_lds_b128(
          (gbl_v4i*)src, (lds_v4i*)dst, 0, 0);
#else
      *(int4*)dst = *(const int4*)src;
#endif
    } else {
      *(int4*)dst = make_int4(0, 0, 0, 0);
    }
  }
#if USE_ASYNC_LDS
  __builtin_amdgcn_s_wait_asynccnt(0);
#endif
  __syncthreads();

  const unsigned char* wrow = wb + (size_t)(cobase + M) * (size_t)kK;
  v8f acc0 = {}, acc1 = {};

  for (int pos = 0; pos < 9; ++pos) {
    const int r = pos / 3, cshift = pos % 3;
#pragma unroll
    for (int cic = 0; cic < 4; ++cic) {
      v16i a;
      const int abase = pos * 512 + cic * 128 + khalf * 16;
#pragma unroll
      for (int half = 0; half < 2; ++half) {
#pragma unroll
        for (int qq = 0; qq < 4; ++qq) {
          const int2 t = *(const int2*)(wrow + abase + half * 64 + (qq >> 1) * 32 + (qq & 1) * 8);
          a[half * 8 + qq * 2] = t.x;
          a[half * 8 + qq * 2 + 1] = t.y;
        }
      }
#pragma unroll
      for (int hx = 0; hx < 2; ++hx) {
        v16i bf;
        const int cell = r * kActCols + (hx * 16 + col + cshift);
        const int bbase = cell * 512 + cic * 128 + khalf * 16;
#pragma unroll
        for (int g = 0; g < 4; ++g) {
          const int4 t = *(const int4*)(act + bbase + g * 32);
          bf[g * 4 + 0] = t.x; bf[g * 4 + 1] = t.y;
          bf[g * 4 + 2] = t.z; bf[g * 4 + 3] = t.w;
        }
        if (hx == 0)
          acc0 = __builtin_amdgcn_wmma_f32_16x16x128_fp8_fp8(a, bf, (short)0, acc0, false, false);
        else
          acc1 = __builtin_amdgcn_wmma_f32_16x16x128_fp8_fp8(a, bf, (short)0, acc1, false, false);
      }
    }
  }
  __syncthreads();

#pragma unroll
  for (int j = 0; j < 8; ++j) {
    const int Mr = j + (khalf << 3);
    const int co = cobase + Mr;
    const size_t base = (((size_t)(b * kCout + co)) << 10) + h * kW;
    const float v0 = acc0[j], v1 = acc1[j];
    convout[base + col] = v0;
    convout[base + 16 + col] = v1;
    atomicAdd(&lsum[wave * 16 + Mr], v0 + v1);
    atomicAdd(&lsq[wave * 16 + Mr], v0 * v0 + v1 * v1);
  }
  __syncthreads();
  if (tid < 128) {
    atomicAdd(&chansum[blockIdx.y * 128 + tid], lsum[tid]);
    atomicAdd(&chansq[blockIdx.y * 128 + tid], lsq[tid]);
  }
}

__global__ void bn_finalize_kernel(const float* __restrict__ sum, const float* __restrict__ sq,
                                   const float* __restrict__ gamma, const float* __restrict__ beta,
                                   float* __restrict__ scale, float* __restrict__ bias) {
  int c = threadIdx.x;
  const float inv = 1.f / (float)kNPix;
  float mean = sum[c] * inv;
  float var = sq[c] * inv - mean * mean;
  float s = gamma[c] * rsqrtf(var + 1e-5f);
  scale[c] = s;
  bias[c] = beta[c] - mean * s;
}

__global__ __launch_bounds__(256) void epilogue_kernel(
    const float* __restrict__ conv, const float* __restrict__ x,
    const float* __restrict__ scale, const float* __restrict__ bias,
    float* __restrict__ out) {
  int idx = blockIdx.x * 256 + threadIdx.x;
  int co = (idx >> 10) & (kCout - 1);
  int b = idx >> 18;
  int hw = idx & (kHWc - 1);
  float v = conv[idx] * scale[co] + bias[co];
  v += x[(((size_t)(b * kCin + co)) << 10) + hw];
  out[idx] = fminf(1.f, fmaxf(-1.f, v));
}

extern "C" void kernel_launch(void* const* d_in, const int* in_sizes, int n_in,
                              void* d_out, int out_size, void* d_ws, size_t ws_size,
                              hipStream_t stream) {
  const float* x     = (const float*)d_in[0];
  const float* Wt    = (const float*)d_in[1];
  const float* gamma = (const float*)d_in[2];
  const float* beta  = (const float*)d_in[3];

  char* ws = (char*)d_ws;
  unsigned char* xb  = (unsigned char*)(ws + OFF_XB);
  unsigned char* wb  = (unsigned char*)(ws + OFF_WB);
  float* conv  = (float*)(ws + OFF_CONV);
  float* sum   = (float*)(ws + OFF_SUM);
  float* sq    = (float*)(ws + OFF_SQ);
  float* scale = (float*)(ws + OFF_SCALE);
  float* bias  = (float*)(ws + OFF_BIAS);
  float* outf  = (float*)d_out;

  bin_x_kernel<<<(int)(kXbBytes / 256), 256, 0, stream>>>(x, xb);
  bin_w_kernel<<<(int)(kWbBytes / 256), 256, 0, stream>>>(Wt, wb);
  zero_stats_kernel<<<1, 512, 0, stream>>>(sum);
  conv_wmma_kernel<<<dim3(kNPix / 32, kCout / 128), 256, 0, stream>>>(xb, wb, conv, sum, sq);
  bn_finalize_kernel<<<1, 256, 0, stream>>>(sum, sq, gamma, beta, scale, bias);
  epilogue_kernel<<<(int)((size_t)kNPix * kCout / 256), 256, 0, stream>>>(conv, x, scale, bias, outf);
}
